// IntersiteModel_16544214024458
// MI455X (gfx1250) — matrix operand order repaired and checked by reading, not compiled, not run on hardware
//
#include <hip/hip_runtime.h>

typedef __attribute__((ext_vector_type(16))) __bf16   v16bf;
typedef __attribute__((ext_vector_type(8)))  float    v8f;
typedef __attribute__((ext_vector_type(8)))  unsigned v8u;

union Frag { v8u u; v16bf b; };

// ---------- constants ----------
#define NTILES 25000            // 400000 / 16
#define WAVES  2                // waves (tiles) per block

// normalization scales folded into weight panels
#define S1_SC   0.05590169943749474f    // 1/sqrt(320)
#define S1_SCD  0.03227486121839514f    // 1/(sqrt(3)*sqrt(320))
#define S1_VC   0.05590169943749474f    // 1/sqrt(320)
#define S1_VCX  0.03952847075210474f    // 1/(sqrt(2)*sqrt(320))
#define S2_SC   0.07216878364870323f    // 1/sqrt(192)
#define S2_SCD  0.04166666666666667f    // 1/(sqrt(3)*sqrt(192)) = 1/24
#define S2_VC   0.0625f                 // 1/sqrt(256)
#define S2_VCX  0.04419417382415922f    // 1/(sqrt(2)*16)
#define S3_SC   0.02209708691207961f    // 1/sqrt(2048)
#define S3_VC   0.01275775102696326f    // 1/(sqrt(3)*sqrt(2048))

// weight-panel chunk bases (1 chunk = one 32x16 bf16 B-fragment = 256 u32)
#define PB_1A 0     // stage1 out0: K=320, N=32  -> 2*10 chunks
#define PB_1B 20    // stage1 out1: K=320, N=16  -> 10
#define PB_2A 30    // stage2 out0: K=192, N=64  -> 4*6
#define PB_2B 54    // stage2 out1: K=256, N=32  -> 2*8
#define PB_3S 70    // stage3 scalar: K=32, N=32 -> 2
#define PB_3V 72    // stage3 vector: K=32, N=32 -> 2
#define NCHUNKS 74

// per-wave LDS layout (float units)
#define LX   0      // x0[m][16], x1[m][8][3]  (40 per m)
#define LY   640
#define LE   1280   // e0[m][4], e1[m][4][3]   (16 per m)
#define LT0  1536   // t0[m][32]
#define LT1  2048   // t1[m][16][3]
#define LU0  2816   // u0[m][64] -> later s[m][32] | g[m][32]
#define LU1  3840   // u1[m][32][3] -> later v
#define LWAVE 5376
#define LH0  0      // H0[m][32]      (reuses dead input region)
#define LH1  512    // H1[m][32][3]

// one-time RNE pack (weights, prep kernel only)
__device__ inline unsigned pk_bf16(float a, float b) {
  unsigned x = __float_as_uint(a), y = __float_as_uint(b);
  x = (x + 0x7FFFu + ((x >> 16) & 1u)) >> 16;
  y = (y + 0x7FFFu + ((y >> 16) & 1u)) >> 16;
  return (x & 0xFFFFu) | (y << 16);
}
// fast truncating pack: one v_perm_b32 (low half = bf16(a), high half = bf16(b))
__device__ inline unsigned pk2(float a, float b) {
  return __builtin_amdgcn_perm(__float_as_uint(b), __float_as_uint(a), 0x07060302u);
}

__device__ inline float dot3(const float* a, const float* b) {
  return a[0]*b[0] + a[1]*b[1] + a[2]*b[2];
}
__device__ inline float crossc(const float* a, const float* b, int i) {
  const int i1 = (i + 1) % 3, i2 = (i + 2) % 3;
  return a[i1]*b[i2] - a[i2]*b[i1];
}

__device__ inline v8f wmma_bf16(const Frag& a, const Frag& b, v8f c) {
  return __builtin_amdgcn_wmma_f32_16x16x32_bf16(false, a.b, false, b.b,
                                                 (short)0, c, false, false);
}
__device__ inline Frag loadB(const uint4* __restrict__ wsp, int chunk, int lane) {
  uint4 q0 = wsp[chunk * 64 + lane * 2];
  uint4 q1 = wsp[chunk * 64 + lane * 2 + 1];
  Frag f;
  f.u = v8u{q0.x, q0.y, q0.z, q0.w, q1.x, q1.y, q1.z, q1.w};
  return f;
}

// Build a 16x32 bf16 A fragment; val(k) gives this lane's row value at panel row
// kbase+k. A-layout K index (per ISA): ((j&3)*2) + ((j>>2)*16) + lh*8 + h.
template <class F>
__device__ inline void buildFrag(Frag& a, int kbase, int lh, F val) {
#pragma unroll
  for (int j = 0; j < 8; ++j) {
    int k0 = kbase + ((j & 3) * 2) + ((j >> 2) * 16) + lh * 8;
    a.u[j] = pk2(val(k0), val(k0 + 1));
  }
}

// =================== prep: weights -> scaled bf16 B-fragment panels ===================
__global__ void iw_prep(const float* __restrict__ W1_00, const float* __restrict__ W1_11_0,
                        const float* __restrict__ W1_01, const float* __restrict__ W1_10,
                        const float* __restrict__ W1_11_1,
                        const float* __restrict__ W2_00, const float* __restrict__ W2_11_0,
                        const float* __restrict__ W2_01, const float* __restrict__ W2_10,
                        const float* __restrict__ W2_11_1,
                        const float* __restrict__ W3_00, const float* __restrict__ W3_11,
                        unsigned* __restrict__ ws) {
  int e = blockIdx.x * 256 + threadIdx.x;
  if (e >= NCHUNKS * 256) return;
  int chunk = e >> 8, r = e & 255;
  int lane = r >> 3, j = r & 7;
  int lh = lane >> 4, n = lane & 15;
  int kk = ((j & 3) * 2) + ((j >> 2) * 16) + lh * 8;   // B-layout (32x16) K index, half 0
  float val[2];
#pragma unroll
  for (int h = 0; h < 2; ++h) {
    int kloc = kk + h;
    float x = 0.f;
    if (chunk < PB_1B) {                                    // stage1 out0
      int c = chunk - PB_1A, nt = c / 10, kb = c % 10, w = nt*16 + n, k = kb*32 + kloc;
      if (k < 256) { int u = k >> 4, v = k & 15; x = W1_00[(u*16 + v)*32 + w] * S1_SC; }
      else { int q = k - 256, u = q >> 3, v = q & 7; x = W1_11_0[(u*8 + v)*32 + w] * S1_SCD; }
    } else if (chunk < PB_2A) {                             // stage1 out1
      int kb = chunk - PB_1B, w = n, k = kb*32 + kloc;
      if (k < 128) { int u = k >> 3, v = k & 7; x = W1_01[(u*8 + v)*16 + w] * S1_VC; }
      else if (k < 256) { int q = k - 128, u = q >> 4, v = q & 15; x = W1_10[(u*16 + v)*16 + w] * S1_VC; }
      else { int q = k - 256, u = q >> 3, v = q & 7; x = W1_11_1[(u*8 + v)*16 + w] * S1_VCX; }
    } else if (chunk < PB_2B) {                             // stage2 out0
      int c = chunk - PB_2A, nt = c / 6, kb = c % 6, w = nt*16 + n, k = kb*32 + kloc;
      if (k < 128) { int u = k >> 2, v = k & 3; x = W2_00[(u*4 + v)*64 + w] * S2_SC; }
      else { int q = k - 128, u = q >> 2, v = q & 3; x = W2_11_0[(u*4 + v)*64 + w] * S2_SCD; }
    } else if (chunk < PB_3S) {                             // stage2 out1
      int c = chunk - PB_2B, nt = c / 8, kb = c % 8, w = nt*16 + n, k = kb*32 + kloc;
      if (k < 128) { int u = k >> 2, v = k & 3; x = W2_01[(u*4 + v)*32 + w] * S2_VC; }
      else if (k < 192) { int q = k - 128, u = q >> 2, v = q & 3; x = W2_10[(u*4 + v)*32 + w] * S2_VC; }
      else { int q = k - 192, u = q >> 2, v = q & 3; x = W2_11_1[(u*4 + v)*32 + w] * S2_VCX; }
    } else if (chunk < PB_3V) {                             // stage3 scalar square
      int nt = chunk - PB_3S, w = nt*16 + n;
      x = W3_00[w*32 + kloc] * S3_SC;
    } else {                                                // stage3 vector square
      int nt = chunk - PB_3V, w = nt*16 + n;
      x = W3_11[w*32 + kloc] * S3_VC;
    }
    val[h] = x;
  }
  ws[e] = pk_bf16(val[0], val[1]);
}

// =================== main: one wave = one 16-element batch tile ===================
__global__ __launch_bounds__(WAVES * 32)
__attribute__((amdgpu_waves_per_eu(4)))
void iw_main(const float* __restrict__ site1, const float* __restrict__ site2,
             const float* __restrict__ edge, const uint4* __restrict__ wsp,
             float* __restrict__ out) {
  __shared__ __align__(16) float lds[WAVES * LWAVE];
  const int lane = threadIdx.x & 31;
  const int wv   = threadIdx.x >> 5;
  const int lh   = lane >> 4;       // lane half
  const int nn   = lane & 15;       // A-row / D-col index
  float* Sm = lds + wv * LWAVE;

  for (int tile = blockIdx.x * WAVES + wv; tile < NTILES; tile += gridDim.x * WAVES) {
    // ---- stage tile inputs into LDS (rows are contiguous in memory) ----
    {
      const float4* a = (const float4*)(site1 + (size_t)tile * 640);
      const float4* b = (const float4*)(site2 + (size_t)tile * 640);
      const float4* c = (const float4*)(edge  + (size_t)tile * 256);
      float4* lx = (float4*)(Sm + LX);
      float4* ly = (float4*)(Sm + LY);
      float4* le = (float4*)(Sm + LE);
#pragma unroll
      for (int t = 0; t < 5; ++t) { lx[lane + 32*t] = a[lane + 32*t];
                                    ly[lane + 32*t] = b[lane + 32*t]; }
#pragma unroll
      for (int t = 0; t < 2; ++t) le[lane + 32*t] = c[lane + 32*t];
    }

    // ---- stage 1: out0 (K=320 = 256 outer + 64 dot, N=32) -> t0 ----
#pragma unroll 1
    for (int nt = 0; nt < 2; ++nt) {
      v8f acc = {0.f,0.f,0.f,0.f,0.f,0.f,0.f,0.f};
#pragma unroll 2
      for (int kb = 0; kb < 8; ++kb) {
        Frag bf = loadB(wsp, PB_1A + nt*10 + kb, lane);
        Frag a;
        buildFrag(a, 32*kb, lh, [&](int k) {
          int u = k >> 4, v = k & 15;
          return Sm[LX + nn*40 + u] * Sm[LY + nn*40 + v]; });
        acc = wmma_bf16(a, bf, acc);
      }
#pragma unroll
      for (int kb = 8; kb < 10; ++kb) {
        Frag bf = loadB(wsp, PB_1A + nt*10 + kb, lane);
        Frag a;
        buildFrag(a, 32*(kb-8), lh, [&](int q) {
          int u = q >> 3, v = q & 7;
          return dot3(&Sm[LX + nn*40 + 16 + u*3], &Sm[LY + nn*40 + 16 + v*3]); });
        acc = wmma_bf16(a, bf, acc);
      }
#pragma unroll
      for (int r = 0; r < 8; ++r)
        Sm[LT0 + (r + 8*lh)*32 + nt*16 + nn] = acc[r];
    }

    // ---- stage 1: out1 per component (K=320 = 128+128+64, N=16) -> t1 ----
#pragma unroll
    for (int i = 0; i < 3; ++i) {
      v8f acc = {0.f,0.f,0.f,0.f,0.f,0.f,0.f,0.f};
#pragma unroll 2
      for (int kb = 0; kb < 4; ++kb) {
        Frag bf = loadB(wsp, PB_1B + kb, lane);
        Frag a;
        buildFrag(a, 32*kb, lh, [&](int k) {
          int u = k >> 3, v = k & 7;
          return Sm[LX + nn*40 + u] * Sm[LY + nn*40 + 16 + v*3 + i]; });
        acc = wmma_bf16(a, bf, acc);
      }
#pragma unroll 2
      for (int kb = 4; kb < 8; ++kb) {
        Frag bf = loadB(wsp, PB_1B + kb, lane);
        Frag a;
        buildFrag(a, 32*(kb-4), lh, [&](int q) {
          int u = q >> 4, v = q & 15;
          return Sm[LX + nn*40 + 16 + u*3 + i] * Sm[LY + nn*40 + v]; });
        acc = wmma_bf16(a, bf, acc);
      }
#pragma unroll
      for (int kb = 8; kb < 10; ++kb) {
        Frag bf = loadB(wsp, PB_1B + kb, lane);
        Frag a;
        buildFrag(a, 32*(kb-8), lh, [&](int q) {
          int u = q >> 3, v = q & 7;
          return crossc(&Sm[LX + nn*40 + 16 + u*3], &Sm[LY + nn*40 + 16 + v*3], i); });
        acc = wmma_bf16(a, bf, acc);
      }
#pragma unroll
      for (int r = 0; r < 8; ++r)
        Sm[LT1 + ((r + 8*lh)*16 + nn)*3 + i] = acc[r];
    }

    // ---- stage 2: out0 (K=192 = 128 outer + 64 dot, N=64) -> u0 ----
#pragma unroll 1
    for (int nt = 0; nt < 4; ++nt) {
      v8f acc = {0.f,0.f,0.f,0.f,0.f,0.f,0.f,0.f};
#pragma unroll 2
      for (int kb = 0; kb < 4; ++kb) {
        Frag bf = loadB(wsp, PB_2A + nt*6 + kb, lane);
        Frag a;
        buildFrag(a, 32*kb, lh, [&](int k) {
          int u = k >> 2, v = k & 3;
          return Sm[LT0 + nn*32 + u] * Sm[LE + nn*16 + v]; });
        acc = wmma_bf16(a, bf, acc);
      }
#pragma unroll
      for (int kb = 4; kb < 6; ++kb) {
        Frag bf = loadB(wsp, PB_2A + nt*6 + kb, lane);
        Frag a;
        buildFrag(a, 32*(kb-4), lh, [&](int q) {
          int u = q >> 2, v = q & 3;
          return dot3(&Sm[LT1 + (nn*16 + u)*3], &Sm[LE + nn*16 + 4 + v*3]); });
        acc = wmma_bf16(a, bf, acc);
      }
#pragma unroll
      for (int r = 0; r < 8; ++r)
        Sm[LU0 + (r + 8*lh)*64 + nt*16 + nn] = acc[r];
    }

    // ---- stage 2: out1 per component (K=256 = 128+64+64, N=32) -> u1 ----
#pragma unroll
    for (int i = 0; i < 3; ++i) {
#pragma unroll 1
      for (int nt = 0; nt < 2; ++nt) {
        v8f acc = {0.f,0.f,0.f,0.f,0.f,0.f,0.f,0.f};
#pragma unroll 2
        for (int kb = 0; kb < 4; ++kb) {
          Frag bf = loadB(wsp, PB_2B + nt*8 + kb, lane);
          Frag a;
          buildFrag(a, 32*kb, lh, [&](int k) {
            int u = k >> 2, v = k & 3;
            return Sm[LT0 + nn*32 + u] * Sm[LE + nn*16 + 4 + v*3 + i]; });
          acc = wmma_bf16(a, bf, acc);
        }
#pragma unroll
        for (int kb = 4; kb < 6; ++kb) {
          Frag bf = loadB(wsp, PB_2B + nt*8 + kb, lane);
          Frag a;
          buildFrag(a, 32*(kb-4), lh, [&](int q) {
            int u = q >> 2, v = q & 3;
            return Sm[LT1 + (nn*16 + u)*3 + i] * Sm[LE + nn*16 + v]; });
          acc = wmma_bf16(a, bf, acc);
        }
#pragma unroll
        for (int kb = 6; kb < 8; ++kb) {
          Frag bf = loadB(wsp, PB_2B + nt*8 + kb, lane);
          Frag a;
          buildFrag(a, 32*(kb-6), lh, [&](int q) {
            int u = q >> 2, v = q & 3;
            return crossc(&Sm[LT1 + (nn*16 + u)*3], &Sm[LE + nn*16 + 4 + v*3], i); });
          acc = wmma_bf16(a, bf, acc);
        }
#pragma unroll
        for (int r = 0; r < 8; ++r)
          Sm[LU1 + ((r + 8*lh)*32 + nt*16 + nn)*3 + i] = acc[r];
      }
    }

    // ---- gate: s = silu(u0[:32]), g = sigmoid(u0[32:]), v = u1*g (in place) ----
#pragma unroll 4
    for (int t = 0; t < 16; ++t) {
      int idx = lane + 32*t, m = idx >> 5, w = idx & 31;
      float a = Sm[LU0 + m*64 + w];
      float b = Sm[LU0 + m*64 + 32 + w];
      float g = 1.f / (1.f + __expf(-b));
      Sm[LU0 + m*64 + w]      = a / (1.f + __expf(-a));
      Sm[LU0 + m*64 + 32 + w] = g;
      Sm[LU1 + (m*32 + w)*3 + 0] *= g;
      Sm[LU1 + (m*32 + w)*3 + 1] *= g;
      Sm[LU1 + (m*32 + w)*3 + 2] *= g;
    }

    // ---- stage 3: H0 = W3_00 . s  (K=32, N=32) ----
#pragma unroll 1
    for (int nt = 0; nt < 2; ++nt) {
      v8f acc = {0.f,0.f,0.f,0.f,0.f,0.f,0.f,0.f};
      Frag bf = loadB(wsp, PB_3S + nt, lane);
      Frag a;
      buildFrag(a, 0, lh, [&](int k) { return Sm[LU0 + nn*64 + k]; });
      acc = wmma_bf16(a, bf, acc);
#pragma unroll
      for (int r = 0; r < 8; ++r)
        Sm[LH0 + (r + 8*lh)*32 + nt*16 + nn] = acc[r];
    }
    // ---- stage 3: H1_i = W3_11 . v_i ----
#pragma unroll
    for (int i = 0; i < 3; ++i) {
#pragma unroll 1
      for (int nt = 0; nt < 2; ++nt) {
        v8f acc = {0.f,0.f,0.f,0.f,0.f,0.f,0.f,0.f};
        Frag bf = loadB(wsp, PB_3V + nt, lane);
        Frag a;
        buildFrag(a, 0, lh, [&](int k) { return Sm[LU1 + (nn*32 + k)*3 + i]; });
        acc = wmma_bf16(a, bf, acc);
#pragma unroll
        for (int r = 0; r < 8; ++r)
          Sm[LH1 + ((r + 8*lh)*32 + nt*16 + nn)*3 + i] = acc[r];
      }
    }

    // ---- final dot: out[b] = s.H0 + sum_i v_i.H1_i (scales folded) ----
    float part = 0.f;
#pragma unroll
    for (int t = 0; t < 16; ++t) {
      int u = lh*16 + t;
      part += Sm[LU0 + nn*64 + u] * Sm[LH0 + nn*32 + u];
      const float* vp = &Sm[LU1 + (nn*32 + u)*3];
      const float* hp = &Sm[LH1 + (nn*32 + u)*3];
      part += vp[0]*hp[0] + vp[1]*hp[1] + vp[2]*hp[2];
    }
    part += __shfl_xor(part, 16, 32);
    if (lane < 16) out[tile*16 + nn] = part;
  }
}

extern "C" void kernel_launch(void* const* d_in, const int* in_sizes, int n_in,
                              void* d_out, int out_size, void* d_ws, size_t ws_size,
                              hipStream_t stream) {
  (void)in_sizes; (void)n_in; (void)out_size; (void)ws_size;
  const float* site1 = (const float*)d_in[0];
  const float* site2 = (const float*)d_in[1];
  const float* edge  = (const float*)d_in[2];

  iw_prep<<<NCHUNKS, 256, 0, stream>>>(
      (const float*)d_in[3],  (const float*)d_in[4],  (const float*)d_in[5],
      (const float*)d_in[6],  (const float*)d_in[7],
      (const float*)d_in[8],  (const float*)d_in[9],  (const float*)d_in[10],
      (const float*)d_in[11], (const float*)d_in[12],
      (const float*)d_in[13], (const float*)d_in[14],
      (unsigned*)d_ws);

  iw_main<<<NTILES / WAVES, WAVES * 32, 0, stream>>>(
      site1, site2, edge, (const uint4*)d_ws, (float*)d_out);
}
